// HoloAttentionV2_50663434224034
// MI455X (gfx1250) — matrix operand order repaired and checked by reading, not compiled, not run on hardware
//
#include <hip/hip_runtime.h>
#include <hip/hip_bf16.h>
#include <math.h>

// ---------------------------------------------------------------------------
// HoloAttentionV2 for gfx1250 (MI455X, wave32, WMMA).
//   B=2, T=4096, DM=1024, H=16, D=64, HD=1024.
// Pipeline:
//   1) split x -> bf16 hi/lo               (bf16x3 precision recovery)
//   2) transpose+split Wk,Wv,Wo -> bf16 [N][K] hi/lo
//   3) WMMA GEMM: kreal = x@Wk, vbuf = x@Wv   (v_wmma_f32_16x16x32_bf16)
//   4) chunked two-pass complex scan over T (pos + assoc paths, gated),
//      written in-place into vbuf
//   5) split vbuf -> bf16 hi/lo (reuse x buffers), WMMA GEMM @Wo -> d_out
// ---------------------------------------------------------------------------

typedef __attribute__((ext_vector_type(16))) __bf16 v16bf;
typedef __attribute__((ext_vector_type(8)))  __bf16 v8bf;
typedef __attribute__((ext_vector_type(8)))  float  v8f;

namespace {
constexpr int   Bc   = 2;
constexpr int   Tc   = 4096;
constexpr int   DMc  = 1024;
constexpr int   Hc   = 16;
constexpr int   Dc   = 64;
constexpr int   HDc  = Hc * Dc;        // 1024
constexpr int   Mrows = Bc * Tc;       // 8192
constexpr int   CHUNK = 128;
constexpr int   NCHUNK = Tc / CHUNK;   // 32
constexpr int   NCH  = Bc * HDc;       // 2048 scan channels
constexpr float PHASE_SCALE = 10.0f;
}

// ---------------------------------------------------------------------------
// WMMA helper: D = A(16x32 bf16) x B(32x16 bf16) + C(16x16 f32)
// ---------------------------------------------------------------------------
static __device__ __forceinline__ v8f wmma_bf16(v16bf a, v16bf b, v8f c) {
  return __builtin_amdgcn_wmma_f32_16x16x32_bf16(
      /*neg_a=*/false, a, /*neg_b=*/false, b,
      /*c_mod=*/(short)0, c, /*reuse_a=*/false, /*reuse_b=*/false);
}

static __device__ __forceinline__ v16bf ld_frag2(const __bf16* p0, const __bf16* p1) {
  v8bf a = *(const v8bf*)p0;
  v8bf b = *(const v8bf*)p1;
  v16bf r;
#pragma unroll
  for (int i = 0; i < 8; ++i) { r[i] = a[i]; r[i + 8] = b[i]; }
  return r;
}

// ---------------------------------------------------------------------------
// Elementwise fp32 -> bf16 (hi) + bf16 residual (lo)
// ---------------------------------------------------------------------------
__global__ void holo_split_kernel(const float* __restrict__ in,
                                  __bf16* __restrict__ hi,
                                  __bf16* __restrict__ lo, int n) {
  int i = blockIdx.x * blockDim.x + threadIdx.x;
  if (i >= n) return;
  float x = in[i];
  __bf16 h = (__bf16)x;
  hi[i] = h;
  lo[i] = (__bf16)(x - (float)h);
}

// Transposed split: W is [K][N] row-major fp32; emit Wt [N][K] bf16 hi/lo.
__global__ void holo_splitT_kernel(const float* __restrict__ W,
                                   __bf16* __restrict__ th,
                                   __bf16* __restrict__ tl, int K, int N) {
  int i = blockIdx.x * blockDim.x + threadIdx.x;   // i = n*K + k
  if (i >= K * N) return;
  int n = i / K;
  int k = i - n * K;
  float x = W[(size_t)k * N + n];
  __bf16 h = (__bf16)x;
  th[i] = h;
  tl[i] = (__bf16)(x - (float)h);
}

// ---------------------------------------------------------------------------
// WMMA GEMM, bf16x3 split precision. C[M][N] = A[M][K] * Bt[N][K]^T.
// One wave computes a 32x32 output block (2x2 WMMA tiles, 3 products each).
// All of M,N,K are multiples of 32. Per-wave control flow is uniform, so
// EXEC is all-ones at every WMMA (ISA requirement).
// ---------------------------------------------------------------------------
__global__ __launch_bounds__(128)
void holo_gemm_bf16x3(const __bf16* __restrict__ Ah, const __bf16* __restrict__ Al,
                      const __bf16* __restrict__ Bth, const __bf16* __restrict__ Btl,
                      float* __restrict__ C, int M, int N, int K) {
  const int lane = threadIdx.x & 31;
  const int wid  = blockIdx.x * (blockDim.x >> 5) + (threadIdx.x >> 5);
  const int ntiles = N >> 5;
  const int tM = wid / ntiles;
  const int tN = wid - tM * ntiles;
  if (tM * 32 >= M) return;                 // wave-uniform guard
  const int rowBase = tM * 32;
  const int colBase = tN * 32;
  const int lm = lane & 15;
  const int hi = lane >> 4;                 // half-wave select

  v8f acc[2][2] = {};

  for (int kk = 0; kk < K; kk += 32) {
    v16bf aH[2], aL[2], bH[2], bL[2];
    // A fragment: lane holds row M=lm; lanes 0-15 K runs {kk..kk+7, kk+16..23},
    // lanes 16-31 K runs {kk+8..15, kk+24..31}  (ISA 16-bit A 16x32 layout)
#pragma unroll
    for (int r = 0; r < 2; ++r) {
      size_t base = (size_t)(rowBase + r * 16 + lm) * K + kk + hi * 8;
      aH[r] = ld_frag2(Ah + base, Ah + base + 16);
      aL[r] = ld_frag2(Al + base, Al + base + 16);
    }
    // B fragment from transposed weights: lane holds col N=lm; same K runs as A
    // ({kk..kk+7, kk+16..23} lanes 0-15, {kk+8..15, kk+24..31} lanes 16-31)
#pragma unroll
    for (int c = 0; c < 2; ++c) {
      size_t base = (size_t)(colBase + c * 16 + lm) * K + kk + hi * 8;
      bH[c] = ld_frag2(Bth + base, Bth + base + 16);
      bL[c] = ld_frag2(Btl + base, Btl + base + 16);
    }
#pragma unroll
    for (int r = 0; r < 2; ++r)
#pragma unroll
      for (int c = 0; c < 2; ++c) {
        acc[r][c] = wmma_bf16(aH[r], bH[c], acc[r][c]);  // hi*hi
        acc[r][c] = wmma_bf16(aH[r], bL[c], acc[r][c]);  // hi*lo
        acc[r][c] = wmma_bf16(aL[r], bH[c], acc[r][c]);  // lo*hi
      }
  }

  // C 16x16 f32 layout: VGPR j -> row j (lanes 0-15) / j+8 (lanes 16-31), N=lane%16
#pragma unroll
  for (int r = 0; r < 2; ++r)
#pragma unroll
    for (int c = 0; c < 2; ++c) {
      int row0 = rowBase + r * 16 + hi * 8;
      int col  = colBase + c * 16 + lm;
#pragma unroll
      for (int j = 0; j < 8; ++j)
        C[(size_t)(row0 + j) * N + col] = acc[r][c][j];
    }
}

// ---------------------------------------------------------------------------
// Scan pass 1: per (channel, time-chunk) complex partial sums of both paths.
//   pos:   sum_t v[t] * exp(+i * t * f)
//   assoc: sum_t v[t] * exp(-i * 10 * kr[t-1])   (t=0 term is zero)
// gid = chunk*NCH + ch so adjacent threads hit adjacent hd -> coalesced.
// ---------------------------------------------------------------------------
__global__ __launch_bounds__(256)
void holo_scan_p1(const float* __restrict__ kr, const float* __restrict__ v,
                  const float* __restrict__ freqs, float4* __restrict__ csum) {
  int gid = blockIdx.x * blockDim.x + threadIdx.x;
  if (gid >= NCH * NCHUNK) return;
  int chunk = gid / NCH;
  int ch    = gid - chunk * NCH;
  int b  = ch / HDc;
  int hd = ch - b * HDc;                 // = h*D + d; freqs index directly
  float f  = freqs[hd];
  int   t0 = chunk * CHUNK;

  const float* vp = v  + (size_t)(b * Tc + t0) * HDc + hd;
  const float* kp = kr + (size_t)(b * Tc + t0) * HDc + hd;

  float rs, rc; sincosf((float)t0 * f, &rs, &rc);   // rotor at chunk start
  float ss, sc; sincosf(f, &ss, &sc);               // per-step rotation
  float cprev = 0.f, sprev = 0.f;                   // phasor of kr[t-1]
  if (t0 > 0) { float kpr = kp[-HDc]; __sincosf(PHASE_SCALE * kpr, &sprev, &cprev); }

  float pr = 0.f, pi = 0.f, ar = 0.f, ai = 0.f;
  for (int i = 0; i < CHUNK; ++i) {
    float vv = vp[(size_t)i * HDc];
    pr += vv * rc;     pi += vv * rs;               // v * rotor
    ar += vv * cprev;  ai -= vv * sprev;            // v * conj(k[t-1])
    float nrc = rc * sc - rs * ss;
    float nrs = rc * ss + rs * sc;
    rc = nrc; rs = nrs;
    float kc = kp[(size_t)i * HDc];
    __sincosf(PHASE_SCALE * kc, &sprev, &cprev);    // becomes next conj source
  }
  csum[(size_t)ch * NCHUNK + chunk] = make_float4(pr, pi, ar, ai);
}

// ---------------------------------------------------------------------------
// Scan pass 2: rescan each chunk with exclusive chunk prefix; probe, gate,
// scale by rsqrt(t+1); write result IN PLACE over v (each element is read
// exactly once by its owning thread before the overwrite).
// ---------------------------------------------------------------------------
__global__ __launch_bounds__(256)
void holo_scan_p2(const float* __restrict__ kr, float* __restrict__ v,
                  const float* __restrict__ freqs, const float* __restrict__ gate,
                  const float4* __restrict__ csum) {
  int gid = blockIdx.x * blockDim.x + threadIdx.x;
  if (gid >= NCH * NCHUNK) return;
  int chunk = gid / NCH;
  int ch    = gid - chunk * NCH;
  int b  = ch / HDc;
  int hd = ch - b * HDc;
  int h  = hd / Dc;
  float f  = freqs[hd];
  float gp = gate[h * 2 + 0];
  float ga = gate[h * 2 + 1];
  int   t0 = chunk * CHUNK;

  // exclusive prefix over preceding chunk sums (<=31 float4 loads, L2 hot)
  float Pr = 0.f, Pi = 0.f, Ar = 0.f, Ai = 0.f;
  const float4* cs = csum + (size_t)ch * NCHUNK;
  for (int j = 0; j < chunk; ++j) {
    float4 s = cs[j];
    Pr += s.x; Pi += s.y; Ar += s.z; Ai += s.w;
  }

  float*       vp = v  + (size_t)(b * Tc + t0) * HDc + hd;
  const float* kp = kr + (size_t)(b * Tc + t0) * HDc + hd;

  float rs, rc; sincosf((float)t0 * f, &rs, &rc);
  float ss, sc; sincosf(f, &ss, &sc);
  float cprev = 0.f, sprev = 0.f;
  if (t0 > 0) { float kpr = kp[-HDc]; __sincosf(PHASE_SCALE * kpr, &sprev, &cprev); }

  for (int i = 0; i < CHUNK; ++i) {
    float vv = vp[(size_t)i * HDc];
    Pr += vv * rc;     Pi += vv * rs;
    Ar += vv * cprev;  Ai -= vv * sprev;
    float op = Pr * rc + Pi * rs;                   // Re(mem_pos * conj(rotor))
    float kc = kp[(size_t)i * HDc];
    float sk, ck; __sincosf(PHASE_SCALE * kc, &sk, &ck);
    float oa = Ar * ck - Ai * sk;                   // Re(mem_assoc * k[t])
    float inv = rsqrtf((float)(t0 + i + 1));
    vp[(size_t)i * HDc] = (gp * op + ga * oa) * inv;
    float nrc = rc * sc - rs * ss;
    float nrs = rc * ss + rs * sc;
    rc = nrc; rs = nrs;
    cprev = ck; sprev = sk;                         // carry k[t] -> conj(k[t-1])
  }
}

// ---------------------------------------------------------------------------
// Host-side orchestration (all launches on `stream`, graph-capture safe).
// ---------------------------------------------------------------------------
extern "C" void kernel_launch(void* const* d_in, const int* in_sizes, int n_in,
                              void* d_out, int out_size, void* d_ws, size_t ws_size,
                              hipStream_t stream) {
  const float* x     = (const float*)d_in[0];
  const float* Wk    = (const float*)d_in[1];
  const float* Wv    = (const float*)d_in[2];
  const float* Wo    = (const float*)d_in[3];
  const float* gate  = (const float*)d_in[4];
  const float* freqs = (const float*)d_in[5];
  float* out = (float*)d_out;

  const size_t NX = (size_t)Mrows * DMc;   // 8,388,608 activations
  const size_t NW = (size_t)DMc * HDc;     // 1,048,576 per weight

  char*  ws  = (char*)d_ws;
  size_t off = 0;
  auto carve = [&](size_t bytes) -> void* {
    void* p = ws + off;
    off += (bytes + 255) & ~(size_t)255;
    return p;
  };
  __bf16* xh   = (__bf16*)carve(NX * 2);   // reused for out_comb split later
  __bf16* xl   = (__bf16*)carve(NX * 2);
  __bf16* wkh  = (__bf16*)carve(NW * 2);
  __bf16* wkl  = (__bf16*)carve(NW * 2);
  __bf16* wvh  = (__bf16*)carve(NW * 2);
  __bf16* wvl  = (__bf16*)carve(NW * 2);
  __bf16* woh  = (__bf16*)carve(NW * 2);
  __bf16* wol  = (__bf16*)carve(NW * 2);
  float*  kreal = (float*)carve(NX * 4);
  float*  vbuf  = (float*)carve(NX * 4);   // becomes gated scan output in place
  float4* csum  = (float4*)carve((size_t)NCH * NCHUNK * sizeof(float4));
  (void)ws_size; (void)in_sizes; (void)n_in; (void)out_size;

  dim3 blk(256);
  holo_split_kernel <<<dim3((unsigned)((NX + 255) / 256)), blk, 0, stream>>>(x, xh, xl, (int)NX);
  holo_splitT_kernel<<<dim3((unsigned)((NW + 255) / 256)), blk, 0, stream>>>(Wk, wkh, wkl, DMc, HDc);
  holo_splitT_kernel<<<dim3((unsigned)((NW + 255) / 256)), blk, 0, stream>>>(Wv, wvh, wvl, DMc, HDc);
  holo_splitT_kernel<<<dim3((unsigned)((NW + 255) / 256)), blk, 0, stream>>>(Wo, woh, wol, HDc, DMc);

  const int gemmWaves = (Mrows / 32) * (HDc / 32);  // 8192 waves
  dim3 gblk(128);                                   // 4 waves / block
  dim3 ggrid((unsigned)(gemmWaves / 4));
  holo_gemm_bf16x3<<<ggrid, gblk, 0, stream>>>(xh, xl, wkh, wkl, kreal, Mrows, HDc, DMc);
  holo_gemm_bf16x3<<<ggrid, gblk, 0, stream>>>(xh, xl, wvh, wvl, vbuf,  Mrows, HDc, DMc);

  dim3 sgrid((unsigned)((NCH * NCHUNK) / 256));     // 256 blocks
  holo_scan_p1<<<sgrid, blk, 0, stream>>>(kreal, vbuf, freqs, csum);
  holo_scan_p2<<<sgrid, blk, 0, stream>>>(kreal, vbuf, freqs, gate, csum);

  holo_split_kernel<<<dim3((unsigned)((NX + 255) / 256)), blk, 0, stream>>>(vbuf, xh, xl, (int)NX);
  holo_gemm_bf16x3<<<ggrid, gblk, 0, stream>>>(xh, xl, woh, wol, out, Mrows, DMc, HDc);
}
